// SPIN_71846212927877
// MI455X (gfx1250) — matrix operand order repaired and checked by reading, not compiled, not run on hardware
//
#include <hip/hip_runtime.h>
#include <hip/hip_bf16.h>

// ---------------------------------------------------------------------------
// CDNA5 (gfx1250, wave32) implementation of a GATv2-style edge-MLP layer.
// GEMMs via v_wmma_f32_16x16x32_f16; edge-MLP weights staged in LDS by the
// Tensor Data Mover (tensor_load_to_lds) and consumed via ds_load_b128.
// ---------------------------------------------------------------------------

typedef __attribute__((ext_vector_type(16))) _Float16 v16h;
typedef __attribute__((ext_vector_type(8)))  float    v8f;
typedef __attribute__((ext_vector_type(4)))  unsigned v4u;
typedef __attribute__((ext_vector_type(8)))  unsigned v8u;

#define DD 128          // C = D = M = 128
#define NT 8            // 128 / 16 column tiles
#define KC 4            // 128 / 32 k-chunks
#define WPB 8           // waves per block (256 threads)

__device__ __forceinline__ v8f wmma32(v16h a, v16h b, v8f c) {
  // 8 args: (neg_a, A, neg_b, B, c_mod, C, reuse_a, reuse_b)
  return __builtin_amdgcn_wmma_f32_16x16x32_f16(
      false, a, false, b, (short)0, c, false, false);
}

// A fragment: 16x32 f16 slice for k-chunk kc from row-major 16x128 halfs.
// K order matches load_b_frag (contiguous):
//   lanes 0-15  : row = lane,    K = 0..15  (of the 32-chunk)
//   lanes 16-31 : row = lane-16, K = 16..31
__device__ __forceinline__ v16h load_a_frag(const _Float16* A, int kc, int lane) {
  int r     = lane & 15;
  int khalf = (lane >> 4) << 4;               // 0 or 16
  const _Float16* p = A + r * DD + kc * 32 + khalf;
  v16h a;
#pragma unroll
  for (int t = 0; t < 16; ++t) a[t] = p[t];
  return a;
}

// B fragment: 32x16 f16 tile for k-chunk kc, column tile nt, from transposed
// weight Wt[n][k] (k contiguous). Contiguous K order (matches load_a_frag):
//   lanes 0-15  : col = lane,    K = kc*32 + 0..15   (VGPR v: K=2v,2v+1)
//   lanes 16-31 : col = lane-16, K = kc*32 + 16..31
// => 16 contiguous halfs per lane (32B contiguous load).
__device__ __forceinline__ v16h load_b_frag(const _Float16* Wt, int kc, int nt, int lane) {
  int n  = nt * 16 + (lane & 15);
  int kb = kc * 32 + ((lane >> 4) << 4);      // 0 or 16
  const _Float16* p = Wt + n * DD + kb;
  v16h b;
#pragma unroll
  for (int t = 0; t < 16; ++t) b[t] = p[t];
  return b;
}

// Issue a TDM 2D tile load: 128x128 f16 (32KB), global (row stride 128 halfs)
// -> LDS. Descriptor bitfields per CDNA5 ISA.
__device__ __forceinline__ void tdm_load_weight(const _Float16* gsrc, unsigned lds_byte_addr) {
  unsigned long long ga = (unsigned long long)(size_t)gsrc;
  v4u g0;
  g0[0] = 1u;                                        // count=1, user mode
  g0[1] = lds_byte_addr;                             // lds_addr[31:0]
  g0[2] = (unsigned)(ga & 0xffffffffu);              // global_addr[31:0]
  g0[3] = (unsigned)((ga >> 32) & 0x01ffffffu)       // global_addr[56:32]
        | 0x80000000u;                               // type=2 ("image")
  v8u g1;
  g1[0] = 1u << 16;                                  // data_size=1 (2 bytes)
  g1[1] = (DD & 0xffffu) << 16;                      // tensor_dim0[15:0]
  g1[2] = (DD >> 16) | ((DD & 0xffffu) << 16);       // dim0 hi | tensor_dim1 lo
  g1[3] = (DD >> 16) | ((unsigned)DD << 16);         // dim1 hi | tile_dim0=128
  g1[4] = (unsigned)DD;                              // tile_dim1=128, tile_dim2=0
  g1[5] = (unsigned)DD;                              // tensor_dim0_stride[31:0]=128
  g1[6] = ((unsigned)(DD * DD) & 0xffffu) << 16;     // stride0 hi=0 | stride1 lo
  g1[7] = (unsigned)(DD * DD) >> 16;                 // stride1 hi
  v4u g2 = {0u, 0u, 0u, 0u};                         // dims 2/3 unused (zero)
  v4u g3 = {0u, 0u, 0u, 0u};
  asm volatile("tensor_load_to_lds %0, %1, %2, %3"
               :: "s"(g0), "s"(g1), "s"(g2), "s"(g3) : "memory");
}

__device__ __forceinline__ void atomicMaxF(float* addr, float val) {
  unsigned int* ua = (unsigned int*)addr;
  unsigned int old = __float_as_uint(*addr);
  while (__uint_as_float(old) < val) {
    unsigned int prev = atomicCAS(ua, old, __float_as_uint(val));
    if (prev == old) break;
    old = prev;
  }
}

// ---------------------------------------------------------------------------
// Small prep kernels
// ---------------------------------------------------------------------------

__global__ void k_init(float* accum, float* gmax, float* gsum, int total, int n) {
  int idx = blockIdx.x * blockDim.x + threadIdx.x;
  if (idx < total) accum[idx] = 0.0f;
  if (idx < n) { gmax[idx] = -3.0e38f; gsum[idx] = 0.0f; }
}

__global__ void k_f32_to_f16(const float* __restrict__ src, _Float16* __restrict__ dst, int n) {
  int idx = blockIdx.x * blockDim.x + threadIdx.x;
  if (idx < n) dst[idx] = (_Float16)src[idx];
}

// 128x128 weight W[k][n] (row-major) -> Wt[n][k] f16
__global__ void k_w_transpose_f16(const float* __restrict__ w, _Float16* __restrict__ wt) {
  int idx = blockIdx.x * blockDim.x + threadIdx.x;   // 16384 elements
  int k = idx >> 7, n = idx & 127;
  wt[n * DD + k] = (_Float16)w[k * DD + n];
}

// ---------------------------------------------------------------------------
// Node projections: msg_src = x@W_src + b_src, msg_tgt = x@W_tgt, skip = x@W_skip
// One wave per 16-row tile. 96 WMMAs per wave; B-fragments preloaded per
// N-tile so the 8 b128 loads clause together ahead of the 4-WMMA chain.
// ---------------------------------------------------------------------------

__global__ void k_node_gemm(const _Float16* __restrict__ xh,
                            const _Float16* __restrict__ wsrc_t,
                            const _Float16* __restrict__ wtgt_t,
                            const _Float16* __restrict__ wskip_t,
                            const float* __restrict__ b_src,
                            float* __restrict__ msg_src,
                            float* __restrict__ msg_tgt,
                            float* __restrict__ skip,
                            int n_tiles) {
  int lane = threadIdx.x & 31;
  int wid  = threadIdx.x >> 5;
  int tile = blockIdx.x * WPB + wid;
  if (tile >= n_tiles) return;                 // wave-uniform exit
  size_t row0 = (size_t)tile * 16;

  v16h a[KC];
#pragma unroll
  for (int kc = 0; kc < KC; ++kc) a[kc] = load_a_frag(xh + row0 * DD, kc, lane);

  const _Float16* wts[3] = { wsrc_t, wtgt_t, wskip_t };
  float*          outs[3] = { msg_src, msg_tgt, skip };

  int ncol = lane & 15;
  int moff = (lane >> 4) << 3;

#pragma unroll
  for (int w = 0; w < 3; ++w) {
#pragma unroll
    for (int nt = 0; nt < NT; ++nt) {
      v16h b[KC];
#pragma unroll
      for (int kc = 0; kc < KC; ++kc) b[kc] = load_b_frag(wts[w], kc, nt, lane);
      v8f acc = {};
#pragma unroll
      for (int kc = 0; kc < KC; ++kc) acc = wmma32(a[kc], b[kc], acc);
      float bias = (w == 0) ? b_src[nt * 16 + ncol] : 0.0f;
#pragma unroll
      for (int r = 0; r < 8; ++r)
        outs[w][(row0 + moff + r) * DD + nt * 16 + ncol] = acc[r] + bias;
    }
  }
}

// ---------------------------------------------------------------------------
// Edge MLP: W1t/W2t DMA'd to LDS by the TDM (overlapped with the per-edge
// gather), then per 16-edge tile (one wave):
//   z = prelu(msg_src[j] + msg_tgt[i], a0)          -> LDS f16 A tile
//   h = prelu(z @ W1 + b1, a1)                      -> LDS f16 A tile (reuse)
//   msg = h @ W2 + b2                               -> global
//   gate = msg @ Wg; atomicMax into per-target gmax
// 64 WMMAs per wave, B-fragments from LDS (ds_load_b128).
// ---------------------------------------------------------------------------

__global__ void k_edge_mlp(const float* __restrict__ msg_src,
                           const float* __restrict__ msg_tgt,
                           const long long* __restrict__ ei,   // [2,E]
                           const _Float16* __restrict__ w1t,
                           const _Float16* __restrict__ w2t,
                           const float* __restrict__ b1,
                           const float* __restrict__ b2,
                           const float* __restrict__ Wg,
                           const float* __restrict__ a0p,
                           const float* __restrict__ a1p,
                           float* __restrict__ d_msg,
                           float* __restrict__ d_gate,
                           float* __restrict__ d_gmax,
                           long long E, int n_tiles) {
  __shared__ _Float16 lds_w1[DD * DD];          // 32KB
  __shared__ _Float16 lds_w2[DD * DD];          // 32KB
  __shared__ _Float16 tiles[WPB][16 * DD];      // 32KB (4KB per wave)

  int lane = threadIdx.x & 31;
  int wid  = threadIdx.x >> 5;
  int tile = blockIdx.x * WPB + wid;
  bool active = (tile < n_tiles);               // wave-uniform
  long long e0 = (long long)tile * 16;
  _Float16* t = tiles[wid];

  // --- wave 0 kicks off the TDM weight DMA (EXEC-independent, async) ------
  if (wid == 0) {
    tdm_load_weight(w1t, (unsigned)(size_t)(void*)lds_w1);
    tdm_load_weight(w2t, (unsigned)(size_t)(void*)lds_w2);
  }

  float a0 = a0p[0];
  float a1 = a1p[0];

  // ---- build prelu(msg_src[j] + msg_tgt[i]) tile (16 edges x 128 feats) ----
  // (overlaps with the in-flight tensor DMA)
  if (active) {
    int f = lane * 4;
#pragma unroll 4
    for (int r = 0; r < 16; ++r) {
      long long e = e0 + r;
      long long j = ei[e];
      long long i = ei[E + e];
      const float4 s = *(const float4*)(msg_src + j * DD + f);
      const float4 g = *(const float4*)(msg_tgt + i * DD + f);
      float z0 = s.x + g.x, z1 = s.y + g.y, z2 = s.z + g.z, z3 = s.w + g.w;
      z0 = z0 >= 0.0f ? z0 : a0 * z0;
      z1 = z1 >= 0.0f ? z1 : a0 * z1;
      z2 = z2 >= 0.0f ? z2 : a0 * z2;
      z3 = z3 >= 0.0f ? z3 : a0 * z3;
      _Float16* q = t + r * DD + f;
      q[0] = (_Float16)z0; q[1] = (_Float16)z1;
      q[2] = (_Float16)z2; q[3] = (_Float16)z3;
    }
  }
  if (wid == 0) __builtin_amdgcn_s_wait_tensorcnt(0);   // TDM done in wave 0
  __syncthreads();                                      // publish LDS to all

  int ncol = lane & 15;
  int moff = (lane >> 4) << 3;

  v8f acc1[NT];
  if (active) {
    v16h a[KC];
#pragma unroll
    for (int kc = 0; kc < KC; ++kc) a[kc] = load_a_frag(t, kc, lane);
    // ---- layer 1: [16x128] @ W1 (B from LDS) ----
#pragma unroll
    for (int nt = 0; nt < NT; ++nt) {
      v16h b[KC];
#pragma unroll
      for (int kc = 0; kc < KC; ++kc) b[kc] = load_b_frag(lds_w1, kc, nt, lane);
      v8f acc = {};
#pragma unroll
      for (int kc = 0; kc < KC; ++kc) acc = wmma32(a[kc], b[kc], acc);
      acc1[nt] = acc;
    }
  }
  __syncthreads();

  // ---- bias + prelu, write h tile back to LDS as f16 ----
  if (active) {
#pragma unroll
    for (int nt = 0; nt < NT; ++nt) {
      float bv = b1[nt * 16 + ncol];
#pragma unroll
      for (int r = 0; r < 8; ++r) {
        float v = acc1[nt][r] + bv;
        v = v >= 0.0f ? v : a1 * v;
        t[(moff + r) * DD + nt * 16 + ncol] = (_Float16)v;
      }
    }
  }
  __syncthreads();

  if (active) {
    v16h a[KC];
#pragma unroll
    for (int kc = 0; kc < KC; ++kc) a[kc] = load_a_frag(t, kc, lane);

    float p[8];
#pragma unroll
    for (int r = 0; r < 8; ++r) p[r] = 0.0f;

    // ---- layer 2: h @ W2 + b2 (B from LDS); fused gate partials vs Wg ----
#pragma unroll
    for (int nt = 0; nt < NT; ++nt) {
      v16h b[KC];
#pragma unroll
      for (int kc = 0; kc < KC; ++kc) b[kc] = load_b_frag(lds_w2, kc, nt, lane);
      v8f acc = {};
#pragma unroll
      for (int kc = 0; kc < KC; ++kc) acc = wmma32(a[kc], b[kc], acc);
      float bv = b2[nt * 16 + ncol];
      float wg = Wg[nt * 16 + ncol];
#pragma unroll
      for (int r = 0; r < 8; ++r) {
        float v = acc[r] + bv;
        d_msg[(e0 + moff + r) * DD + nt * 16 + ncol] = v;
        p[r] += v * wg;
      }
    }

    // half-wave (16-lane) reduction: row M = r + moff lives on one lane group
#pragma unroll
    for (int r = 0; r < 8; ++r) {
#pragma unroll
      for (int o = 8; o > 0; o >>= 1) p[r] += __shfl_xor(p[r], o, 16);
    }
    if (ncol == 0) {
#pragma unroll
      for (int r = 0; r < 8; ++r) {
        long long e = e0 + moff + r;
        d_gate[e] = p[r];
        long long i = ei[E + e];
        atomicMaxF(&d_gmax[i], p[r]);
      }
    }
  }
}

// ---------------------------------------------------------------------------
// Softmax numerator + segment sum
// ---------------------------------------------------------------------------

__global__ void k_softmax_num(const float* __restrict__ gate,
                              const long long* __restrict__ ei,
                              const float* __restrict__ gmax,
                              float* __restrict__ eg,
                              float* __restrict__ gsum, int E) {
  int e = blockIdx.x * blockDim.x + threadIdx.x;
  if (e >= E) return;
  long long i = ei[(long long)E + e];
  float v = __expf(gate[e] - gmax[i]);
  eg[e] = v;
  atomicAdd(&gsum[i], v);
}

// ---------------------------------------------------------------------------
// alpha-weighted scatter-add aggregation
// ---------------------------------------------------------------------------

__global__ void k_aggregate(const float* __restrict__ msg,
                            const float* __restrict__ eg,
                            const float* __restrict__ gsum,
                            const long long* __restrict__ ei,
                            float* __restrict__ accum, int E) {
  long long idx = (long long)blockIdx.x * blockDim.x + threadIdx.x;
  if (idx >= (long long)E * DD) return;
  int e = (int)(idx >> 7);
  int f = (int)(idx & 127);
  long long i = ei[(long long)E + e];
  float alpha = eg[e] / (gsum[i] + 1e-16f);
  atomicAdd(&accum[i * DD + f], alpha * msg[idx]);
}

// ---------------------------------------------------------------------------
// skip add + LayerNorm (one 128-thread block per row)
// ---------------------------------------------------------------------------

__global__ void k_finalize(const float* __restrict__ accum,
                           const float* __restrict__ skip,
                           const float* __restrict__ gamma,
                           const float* __restrict__ beta,
                           float* __restrict__ out) {
  __shared__ float sm[DD];
  int row = blockIdx.x;
  int f = threadIdx.x;
  float v = accum[(size_t)row * DD + f] + skip[(size_t)row * DD + f];
  sm[f] = v;
  __syncthreads();
  for (int s = 64; s > 0; s >>= 1) { if (f < s) sm[f] += sm[f + s]; __syncthreads(); }
  float mu = sm[0] * (1.0f / DD);
  __syncthreads();
  float d = v - mu;
  sm[f] = d * d;
  __syncthreads();
  for (int s = 64; s > 0; s >>= 1) { if (f < s) sm[f] += sm[f + s]; __syncthreads(); }
  float var = sm[0] * (1.0f / DD);
  out[(size_t)row * DD + f] = d * rsqrtf(var + 1e-5f) * gamma[f] + beta[f];
}

// ---------------------------------------------------------------------------
// Host launcher
// ---------------------------------------------------------------------------

extern "C" void kernel_launch(void* const* d_in, const int* in_sizes, int n_in,
                              void* d_out, int out_size, void* d_ws, size_t ws_size,
                              hipStream_t stream) {
  const float*     x      = (const float*)d_in[0];
  const long long* ei     = (const long long*)d_in[1];   // int64 [2,E]
  const float*     W_src  = (const float*)d_in[2];
  const float*     b_src  = (const float*)d_in[3];
  const float*     W_tgt  = (const float*)d_in[4];
  const float*     W_skip = (const float*)d_in[5];
  const float*     a0     = (const float*)d_in[6];
  const float*     W1     = (const float*)d_in[7];
  const float*     b1     = (const float*)d_in[8];
  const float*     a1     = (const float*)d_in[9];
  const float*     W2     = (const float*)d_in[10];
  const float*     b2     = (const float*)d_in[11];
  const float*     Wg     = (const float*)d_in[12];
  const float*     gamma  = (const float*)d_in[13];
  const float*     beta   = (const float*)d_in[14];

  const int N = in_sizes[0] / DD;
  const int E = in_sizes[1] / 2;

  // workspace carve-out (256B aligned)
  size_t off = 0;
  auto carve = [&](size_t bytes) -> char* {
    char* p = (char*)d_ws + off;
    off += (bytes + 255) & ~(size_t)255;
    return p;
  };
  _Float16* xh      = (_Float16*)carve((size_t)N * DD * 2);
  _Float16* wsrc_t  = (_Float16*)carve(DD * DD * 2);
  _Float16* wtgt_t  = (_Float16*)carve(DD * DD * 2);
  _Float16* wskip_t = (_Float16*)carve(DD * DD * 2);
  _Float16* w1t     = (_Float16*)carve(DD * DD * 2);
  _Float16* w2t     = (_Float16*)carve(DD * DD * 2);
  float* msg_src    = (float*)carve((size_t)N * DD * 4);
  float* msg_tgt    = (float*)carve((size_t)N * DD * 4);
  float* skip       = (float*)carve((size_t)N * DD * 4);
  float* d_msg      = (float*)carve((size_t)E * DD * 4);
  float* d_gate     = (float*)carve((size_t)E * 4);
  float* d_eg       = (float*)carve((size_t)E * 4);
  float* d_gmax     = (float*)carve((size_t)N * 4);
  float* d_gsum     = (float*)carve((size_t)N * 4);
  float* d_accum    = (float*)carve((size_t)N * DD * 4);

  const int nodeElems = N * DD;

  // 1) init accumulators (must happen every call; harness doesn't re-zero)
  k_init<<<(nodeElems + 255) / 256, 256, 0, stream>>>(d_accum, d_gmax, d_gsum, nodeElems, N);

  // 2) precision conversion / weight transposition
  k_f32_to_f16<<<(nodeElems + 255) / 256, 256, 0, stream>>>(x, xh, nodeElems);
  k_w_transpose_f16<<<64, 256, 0, stream>>>(W_src,  wsrc_t);
  k_w_transpose_f16<<<64, 256, 0, stream>>>(W_tgt,  wtgt_t);
  k_w_transpose_f16<<<64, 256, 0, stream>>>(W_skip, wskip_t);
  k_w_transpose_f16<<<64, 256, 0, stream>>>(W1,     w1t);
  k_w_transpose_f16<<<64, 256, 0, stream>>>(W2,     w2t);

  // 3) node projections (WMMA)
  int node_tiles = (N + 15) / 16;                      // 1250
  k_node_gemm<<<(node_tiles + WPB - 1) / WPB, WPB * 32, 0, stream>>>(
      xh, wsrc_t, wtgt_t, wskip_t, b_src, msg_src, msg_tgt, skip, node_tiles);

  // 4) edge MLP + gate + segment-max (WMMA + TDM weight staging)
  int edge_tiles = (E + 15) / 16;                      // 40000
  k_edge_mlp<<<(edge_tiles + WPB - 1) / WPB, WPB * 32, 0, stream>>>(
      msg_src, msg_tgt, ei, w1t, w2t, b1, b2, Wg, a0, a1,
      d_msg, d_gate, d_gmax, (long long)E, edge_tiles);

  // 5) softmax numerator + segment sum
  k_softmax_num<<<(E + 255) / 256, 256, 0, stream>>>(d_gate, ei, d_gmax, d_eg, d_gsum, E);

  // 6) weighted aggregation
  long long aggElems = (long long)E * DD;
  k_aggregate<<<(int)((aggElems + 255) / 256), 256, 0, stream>>>(
      d_msg, d_eg, d_gsum, ei, d_accum, E);

  // 7) skip + LayerNorm
  k_finalize<<<N, DD, 0, stream>>>(d_accum, skip, gamma, beta, (float*)d_out);
}
